// MoEModel_74071005987145
// MI455X (gfx1250) — matrix operand order repaired and checked by reading, not compiled, not run on hardware
//
#include <hip/hip_runtime.h>

typedef __attribute__((ext_vector_type(16))) _Float16 v16h;
typedef __attribute__((ext_vector_type(8)))  _Float16 v8h;
typedef __attribute__((ext_vector_type(4)))  _Float16 v4h;
typedef __attribute__((ext_vector_type(8)))  float    v8f;
typedef __attribute__((ext_vector_type(4)))  int      v4i;

#ifndef USE_ASYNC_LDS
#define USE_ASYNC_LDS 0
#endif
#if defined(__has_builtin)
#if __has_builtin(__builtin_amdgcn_global_load_async_to_lds_b128) && \
    __has_builtin(__builtin_amdgcn_s_wait_asynccnt)
#undef USE_ASYNC_LDS
#define USE_ASYNC_LDS 1
#endif
#endif

static __device__ __forceinline__ v16h join8(v8h a, v8h b)
{
    return __builtin_shufflevector(a, b, 0, 1, 2, 3, 4, 5, 6, 7,
                                   8, 9, 10, 11, 12, 13, 14, 15);
}

// ---------------------------------------------------------------------------
// B=512, x[512,3,32,32], E=4, NC=100, top_k=1
// d_out: final_output [512,100] | router_probs [512,4] | aux_loss [1]
// ---------------------------------------------------------------------------

// ---------------- router: conv(3->16) relu GAP -> fc(16->4) softmax top1 ----
__global__ void router_kernel(const float* __restrict__ x,
                              const float* __restrict__ rcw,   // [16,3,3,3]
                              const float* __restrict__ rcb,   // [16]
                              const float* __restrict__ rfw,   // [4,16]
                              const float* __restrict__ rfb,   // [4]
                              float* __restrict__ probs_out,   // [512,4]
                              int*   __restrict__ idx_ws,      // [512]
                              float* __restrict__ gate_ws,     // [512]
                              int*   __restrict__ counts,      // [4]
                              int*   __restrict__ lists)       // [4,512]
{
    __shared__ float xp[3 * 34 * 34];
    __shared__ float w[16 * 27];
    __shared__ float gsum[16];
    const int b = blockIdx.x, t = threadIdx.x;

    for (int i = t; i < 3 * 34 * 34; i += 256) xp[i] = 0.f;
    for (int i = t; i < 432; i += 256) w[i] = rcw[i];
    if (t < 16) gsum[t] = 0.f;
    __syncthreads();
    for (int i = t; i < 3072; i += 256) {
        int ic = i >> 10, rem = i & 1023, r = rem >> 5, c = rem & 31;
        xp[ic * 1156 + (r + 1) * 34 + (c + 1)] = x[b * 3072 + i];
    }
    __syncthreads();

    float tsum[16];
    for (int ch = 0; ch < 16; ++ch) tsum[ch] = 0.f;
    for (int p = t; p < 1024; p += 256) {
        int r = p >> 5, c = p & 31;
        float acc[16];
        for (int ch = 0; ch < 16; ++ch) acc[ch] = rcb[ch];
        for (int ic = 0; ic < 3; ++ic)
            for (int kh = 0; kh < 3; ++kh)
                for (int kw = 0; kw < 3; ++kw) {
                    float xv = xp[ic * 1156 + (r + kh) * 34 + (c + kw)];
                    int kk = ic * 9 + kh * 3 + kw;
                    for (int ch = 0; ch < 16; ++ch)
                        acc[ch] = fmaf(xv, w[ch * 27 + kk], acc[ch]);
                }
        for (int ch = 0; ch < 16; ++ch) tsum[ch] += fmaxf(acc[ch], 0.f);
    }
    for (int ch = 0; ch < 16; ++ch) atomicAdd(&gsum[ch], tsum[ch]);
    __syncthreads();

    if (t == 0) {
        float g[16], l[4];
        for (int ch = 0; ch < 16; ++ch) g[ch] = gsum[ch] * (1.0f / 1024.f);
        float mx = -1e30f;
        for (int e = 0; e < 4; ++e) {
            float s = rfb[e];
            for (int ch = 0; ch < 16; ++ch) s = fmaf(g[ch], rfw[e * 16 + ch], s);
            l[e] = s; mx = fmaxf(mx, s);
        }
        float den = 0.f;
        for (int e = 0; e < 4; ++e) { l[e] = __expf(l[e] - mx); den += l[e]; }
        float inv = 1.f / den;
        int arg = 0; float best = -1.f;
        for (int e = 0; e < 4; ++e) {
            float pr = l[e] * inv;
            probs_out[b * 4 + e] = pr;
            if (pr > best) { best = pr; arg = e; }
        }
        idx_ws[b] = arg;
        gate_ws[b] = best;
        int pos = atomicAdd(&counts[arg], 1);
        lists[arg * 512 + pos] = b;
    }
}

// ---- pre-convert e_fc_w to f16 once (L2-resident for the FC GEMM) ---------
__global__ void cvt_fcw_kernel(const float* __restrict__ src,
                               _Float16* __restrict__ dst)   // 4*100*4096
{
    int i = (blockIdx.x * 256 + threadIdx.x) * 4;
    float4 v = *(const float4*)(src + i);
    v4h h;
    h[0] = (_Float16)v.x; h[1] = (_Float16)v.y;
    h[2] = (_Float16)v.z; h[3] = (_Float16)v.w;
    *(v4h*)(dst + i) = h;
}

// ------------- expert conv1: 3->32, relu, maxpool2 (WMMA) ------------------
// Implicit GEMM: M=1024 pixels, N=32 oc, K=27 pad 32. A tile built once per
// (row-pair, col-half), reused by both N-tiles (compile-time nt -> no movrels).
__global__ void conv1_kernel(const float* __restrict__ x,
                             const float* __restrict__ w1,   // [4,32,3,3,3]
                             const float* __restrict__ b1,   // [4,32]
                             const int*   __restrict__ idx_ws,
                             _Float16* __restrict__ h1h)     // [512,16,16,32]
{
    __shared__ __align__(32) _Float16 xp[3472];      // 3*34*34 = 3468, padded
    __shared__ __align__(32) _Float16 wBt[32 * 32];  // [n][k] (K-contiguous)
    __shared__ float bias[32];
    const int b = blockIdx.x, t = threadIdx.x;
    const int e = idx_ws[b];

    {   // zero pad region with 128-bit stores
        uint4 z = make_uint4(0u, 0u, 0u, 0u);
        uint4* zp = (uint4*)xp;
        for (int i = t; i < 434; i += 256) zp[i] = z;
    }
    for (int i = t; i < 1024; i += 256) {
        int n = i >> 5, k = i & 31;
        wBt[i] = (_Float16)((k < 27) ? w1[e * 864 + n * 27 + k] : 0.f);
    }
    if (t < 32) bias[t] = b1[e * 32 + t];
    __syncthreads();
    for (int i = t; i < 3072; i += 256) {
        int ic = i >> 10, rem = i & 1023, r = rem >> 5, c = rem & 31;
        xp[ic * 1156 + (r + 1) * 34 + (c + 1)] = (_Float16)x[b * 3072 + i];
    }
    __syncthreads();

    const int lane = t & 31, wave = t >> 5;
    const int half = lane >> 4, mn = lane & 15;

    // B fragments: K-contiguous -> single 32B LDS vector load per N-tile.
    v16h Bf[2];
#pragma unroll
    for (int nt = 0; nt < 2; ++nt)
        Bf[nt] = *(const v16h*)&wBt[(mn + 16 * nt) * 32 + 16 * half];

    // 16 row-pairs over 8 waves; per (row-pair, col-half): one A pair, 2 N-tiles.
#pragma unroll
    for (int rpi = 0; rpi < 2; ++rpi) {
        int rp = wave * 2 + rpi;
        int r0 = 2 * rp;
#pragma unroll
        for (int colh = 0; colh < 2; ++colh) {
            int col = colh * 16 + mn;
            v16h A0, A1;   // patch gather (K=27, irregular; small share of work)
#pragma unroll
            for (int i = 0; i < 16; ++i) {
                int k = i + 16 * half;
                _Float16 v0 = (_Float16)0.f, v1 = (_Float16)0.f;
                if (k < 27) {
                    int ic = k / 9, rem = k % 9, kh = rem / 3, kw = rem % 3;
                    v0 = xp[ic * 1156 + (r0 + kh) * 34 + (col + kw)];
                    v1 = xp[ic * 1156 + (r0 + 1 + kh) * 34 + (col + kw)];
                }
                A0[i] = v0; A1[i] = v1;
            }
#pragma unroll
            for (int nt = 0; nt < 2; ++nt) {
                int ch = mn + 16 * nt;
                float bv = bias[ch];
                v8f acc0, acc1;
#pragma unroll
                for (int r = 0; r < 8; ++r) { acc0[r] = bv; acc1[r] = bv; }
                acc0 = __builtin_amdgcn_wmma_f32_16x16x32_f16(
                           false, A0, false, Bf[nt], (short)0, acc0, false, false);
                acc1 = __builtin_amdgcn_wmma_f32_16x16x32_f16(
                           false, A1, false, Bf[nt], (short)0, acc1, false, false);
#pragma unroll
                for (int p = 0; p < 4; ++p) {
                    float pv = fmaxf(fmaxf(acc0[2 * p], acc0[2 * p + 1]),
                                     fmaxf(acc1[2 * p], acc1[2 * p + 1]));
                    pv = fmaxf(pv, 0.f);
                    int pc = colh * 8 + half * 4 + p;
                    h1h[((b * 16 + rp) * 16 + pc) * 32 + ch] = (_Float16)pv;
                }
            }
        }
    }
}

// ------------- expert conv2: 32->64, relu, maxpool2 (WMMA) -----------------
// M=256 pixels, N=64 oc, K=288 = 9 taps x 32 ic. One row-pair per wave; A pair
// built once per k-step and fed to 4 N-tile accumulators (independent WMMAs
// hide the WMMA hazard NOPs). Activation staging via async-to-LDS.
__global__ void conv2_kernel(const _Float16* __restrict__ h1h, // [512,16,16,32]
                             const float* __restrict__ w2,     // [4,64,32,3,3]
                             const float* __restrict__ b2,     // [4,64]
                             const int*   __restrict__ idx_ws,
                             _Float16* __restrict__ h2h)       // [512,4096]
{
    __shared__ __align__(32) _Float16 hp2[18 * 18 * 32];   // 10368
    __shared__ __align__(32) _Float16 wB2t[9 * 64 * 32];   // [tap][oc][ic]
    __shared__ float bias[64];
    const int b = blockIdx.x, t = threadIdx.x;
    const int e = idx_ws[b];

    {   // zero-fill padded activation tile (1296 x 16B)
        uint4 z = make_uint4(0u, 0u, 0u, 0u);
        uint4* zp = (uint4*)hp2;
        for (int i = t; i < 1296; i += 256) zp[i] = z;
    }
    for (int i = t; i < 18432; i += 256) {
        int s = i >> 11, rem = i & 2047, ch = rem >> 5, ic = rem & 31;
        wB2t[i] = (_Float16)w2[e * 18432 + ch * 288 + ic * 9 + s];
    }
    if (t < 64) bias[t] = b2[e * 64 + t];
    __syncthreads();

#if USE_ASYNC_LDS
    // 1024 x 16-byte async copies global(f16)->LDS, ASYNCcnt-tracked.
    for (int c2 = t; c2 < 1024; c2 += 256) {
        int pix = c2 >> 2, q = c2 & 3;
        int r = pix >> 4, c = pix & 15;
        const _Float16* gp = h1h + ((size_t)b * 256 + pix) * 32 + q * 8;
        _Float16* lp = &hp2[((r + 1) * 18 + (c + 1)) * 32 + q * 8];
        __builtin_amdgcn_global_load_async_to_lds_b128(
            (__attribute__((address_space(1))) v4i*)gp,
            (__attribute__((address_space(3))) v4i*)lp, 0, 0);
    }
    __builtin_amdgcn_s_wait_asynccnt(0);
#else
    for (int i = t; i < 1024; i += 256) {
        int pix = i >> 2, q = i & 3;
        int r = pix >> 4, c = pix & 15;
        *(v8h*)&hp2[((r + 1) * 18 + (c + 1)) * 32 + q * 8] =
            *(const v8h*)(h1h + ((size_t)b * 256 + pix) * 32 + q * 8);
    }
#endif
    __syncthreads();

    const int lane = t & 31, wave = t >> 5;
    const int half = lane >> 4, mn = lane & 15;

    const int rp = wave;           // 8 row-pairs over 8 waves
    const int r0 = 2 * rp, col = mn;

    v8f acc0[4], acc1[4];
#pragma unroll
    for (int nt = 0; nt < 4; ++nt) {
        float bv = bias[mn + 16 * nt];
#pragma unroll
        for (int r = 0; r < 8; ++r) { acc0[nt][r] = bv; acc1[nt][r] = bv; }
    }

    for (int s = 0; s < 9; ++s) {
        int kh = s / 3, kw = s % 3;
        int p0 = ((r0 + kh) * 18 + (col + kw)) * 32;
        int p1 = ((r0 + 1 + kh) * 18 + (col + kw)) * 32;
        v8h a0lo = *(const v8h*)&hp2[p0 + 8 * half];
        v8h a0hi = *(const v8h*)&hp2[p0 + 16 + 8 * half];
        v8h a1lo = *(const v8h*)&hp2[p1 + 8 * half];
        v8h a1hi = *(const v8h*)&hp2[p1 + 16 + 8 * half];
        v16h A0 = join8(a0lo, a0hi);
        v16h A1 = join8(a1lo, a1hi);
#pragma unroll
        for (int nt = 0; nt < 4; ++nt) {
            v16h Bfr = join8(*(const v8h*)&wB2t[s * 2048 + (mn + 16 * nt) * 32 + 8 * half], *(const v8h*)&wB2t[s * 2048 + (mn + 16 * nt) * 32 + 16 + 8 * half]);
            acc0[nt] = __builtin_amdgcn_wmma_f32_16x16x32_f16(
                           false, A0, false, Bfr, (short)0, acc0[nt], false, false);
            acc1[nt] = __builtin_amdgcn_wmma_f32_16x16x32_f16(
                           false, A1, false, Bfr, (short)0, acc1[nt], false, false);
        }
    }
#pragma unroll
    for (int nt = 0; nt < 4; ++nt) {
        int ch = mn + 16 * nt;
#pragma unroll
        for (int p = 0; p < 4; ++p) {
            float pv = fmaxf(fmaxf(acc0[nt][2 * p], acc0[nt][2 * p + 1]),
                             fmaxf(acc1[nt][2 * p], acc1[nt][2 * p + 1]));
            pv = fmaxf(pv, 0.f);
            int pc = half * 4 + p;
            h2h[(size_t)b * 4096 + ch * 64 + rp * 8 + pc] = (_Float16)pv;
        }
    }
}

// ------------- expert FC (4096->100) over routed sample lists (WMMA) -------
// grid = (32 M-tiles, 4 experts), 224 threads = 7 waves, one N-tile per wave.
// A (f16 h2) and B (pre-converted f16 weights) are K-contiguous -> b128 loads.
__global__ void fc_kernel(const _Float16* __restrict__ h2h,   // [512,4096]
                          const _Float16* __restrict__ fwh,   // [4,100,4096] f16
                          const float* __restrict__ fcb,      // [4,100]
                          const int*   __restrict__ counts,
                          const int*   __restrict__ lists,
                          const float* __restrict__ gate_ws,
                          float* __restrict__ out)             // [512,100]
{
    const int e = blockIdx.y, mt = blockIdx.x;
    __shared__ int   slist[16];
    __shared__ float sgate[16];
    __shared__ int   scount;
    const int t = threadIdx.x;
    if (t == 0) scount = counts[e];
    __syncthreads();
    const int count = scount;
    if (mt * 16 >= count) return;
    if (t < 16) {
        int g = mt * 16 + t;
        int s = (g < count) ? lists[e * 512 + g] : 0;
        slist[t] = s;
        sgate[t] = (g < count) ? gate_ws[s] : 0.f;
    }
    __syncthreads();

    const int lane = t & 31, wave = t >> 5;     // waves 0..6 = N-tiles
    const int half = lane >> 4, mn = lane & 15;
    const int c = wave * 16 + mn;
    const bool cvalid = (c < 100);
    // Invalid rows/cols only pollute D entries that are never stored.
    const _Float16* __restrict__ arow = h2h + (size_t)slist[mn] * 4096;
    const _Float16* __restrict__ brow =
        fwh + (size_t)e * 409600 + (size_t)(cvalid ? c : 0) * 4096;

    v8f acc;
#pragma unroll
    for (int r = 0; r < 8; ++r) acc[r] = 0.f;
    for (int ks = 0; ks < 128; ++ks) {
        __builtin_prefetch(brow + (ks + 4) * 32, 0, 1);
        v8h alo = *(const v8h*)&arow[ks * 32 + 8 * half];
        v8h ahi = *(const v8h*)&arow[ks * 32 + 16 + 8 * half];
        v16h A = join8(alo, ahi);
        v16h Bf = join8(*(const v8h*)&brow[ks * 32 + 8 * half], *(const v8h*)&brow[ks * 32 + 16 + 8 * half]);
        acc = __builtin_amdgcn_wmma_f32_16x16x32_f16(false, A, false, Bf,
                                                     (short)0, acc, false, false);
    }
    if (cvalid) {
        float bb = fcb[e * 100 + c];
#pragma unroll
        for (int r = 0; r < 8; ++r) {
            int m2 = r + 8 * half;
            if (mt * 16 + m2 < count) {
                int s2 = slist[m2];
                out[(size_t)s2 * 100 + c] = sgate[m2] * (acc[r] + bb);
            }
        }
    }
}

// ------------- aux loss: mean over batch, MSE vs 1/E -----------------------
__global__ void aux_kernel(const float* __restrict__ probs, float* __restrict__ auxout)
{
    __shared__ float ssum[4];
    const int t = threadIdx.x;
    if (t < 4) ssum[t] = 0.f;
    __syncthreads();
    float loc[4] = {0.f, 0.f, 0.f, 0.f};
    for (int i = t; i < 512; i += 128)
        for (int e = 0; e < 4; ++e) loc[e] += probs[i * 4 + e];
    for (int e = 0; e < 4; ++e) atomicAdd(&ssum[e], loc[e]);
    __syncthreads();
    if (t == 0) {
        float a = 0.f;
        for (int e = 0; e < 4; ++e) {
            float m = ssum[e] * (1.f / 512.f) - 0.25f;
            a += m * m;
        }
        auxout[0] = a * 0.25f;
    }
}

__global__ void init_kernel(int* counts)
{
    if (threadIdx.x < 4) counts[threadIdx.x] = 0;
}

// ---------------------------------------------------------------------------
extern "C" void kernel_launch(void* const* d_in, const int* in_sizes, int n_in,
                              void* d_out, int out_size, void* d_ws, size_t ws_size,
                              hipStream_t stream)
{
    (void)in_sizes; (void)n_in; (void)out_size; (void)ws_size;
    const float* x   = (const float*)d_in[0];
    const float* rcw = (const float*)d_in[1];
    const float* rcb = (const float*)d_in[2];
    const float* rfw = (const float*)d_in[3];
    const float* rfb = (const float*)d_in[4];
    const float* ew1 = (const float*)d_in[5];
    const float* eb1 = (const float*)d_in[6];
    const float* ew2 = (const float*)d_in[7];
    const float* eb2 = (const float*)d_in[8];
    const float* efw = (const float*)d_in[9];
    const float* efb = (const float*)d_in[10];
    // d_in[11] = top_k (assumed 1, as in setup_inputs)

    float* out   = (float*)d_out;            // [512*100]
    float* probs = out + 512 * 100;          // [512*4]
    float* aux   = out + 512 * 100 + 512 * 4;

    char* ws = (char*)d_ws;
    _Float16* h1h  = (_Float16*)(ws);                  // 512*16*16*32 f16 = 8 MiB
    _Float16* h2h  = (_Float16*)(ws + 8388608);        // 512*4096 f16 = 4 MiB
    _Float16* fwh  = (_Float16*)(ws + 12582912);       // 4*100*4096 f16 = 3.125 MiB
    int*   idx     = (int*)  (ws + 15859712);          // [512]
    float* gate    = (float*)(ws + 15861760);          // [512]
    int*   counts  = (int*)  (ws + 15863808);          // [4]
    int*   lists   = (int*)  (ws + 15863824);          // [4,512]

    init_kernel<<<1, 32, 0, stream>>>(counts);
    cvt_fcw_kernel<<<1600, 256, 0, stream>>>(efw, fwh);  // 1,638,400 elems
    router_kernel<<<512, 256, 0, stream>>>(x, rcw, rcb, rfw, rfb,
                                           probs, idx, gate, counts, lists);
    conv1_kernel<<<512, 256, 0, stream>>>(x, ew1, eb1, idx, h1h);
    conv2_kernel<<<512, 256, 0, stream>>>(h1h, ew2, eb2, idx, h2h);
    fc_kernel<<<dim3(32, 4), 224, 0, stream>>>(h2h, fwh, efb, counts, lists, gate, out);
    aux_kernel<<<1, 128, 0, stream>>>(probs, aux);
}
